// TorchLMHeadKTO_46815143526661
// MI455X (gfx1250) — matrix operand order repaired and checked by reading, not compiled, not run on hardware
//
#include <hip/hip_runtime.h>
#include <cmath>

// ---------------------------------------------------------------------------
// KTO loss for an LM head, MI455X (gfx1250, wave32, WMMA).
//   logits = x @ W^T  (B*T=2048 tokens, H=1024, V=32000)
//   per-token logp = logit[label] - logsumexp_V(logits)
//
// bf16x3-split GEMM on V_WMMA_F32_16X16X32_BF16.
//  - W / ref_W and x / ref_x pre-split once into bf16 hi/lo arrays (x already
//    in A-fragment order): ZERO conversion/swizzle math in the hot kernel.
//  - 64-token A super-tile copied into 256 KB dynamic LDS (CDNA5: 320 KB/WGP)
//    via GLOBAL_LOAD_ASYNC_TO_LDS_B128 (ASYNCcnt) when available.
//  - Waves = 4 token-tiles x 4 v-groups: 4 waves per v-group issue identical
//    W addresses -> WGP$ dedups them (L2 B-traffic 33.6 GB -> 8.4 GB).
//  - Per-lane online logsumexp; single cross-lane (M,S) merge per wave.
// ---------------------------------------------------------------------------

typedef __attribute__((ext_vector_type(16))) __bf16 v16bf;
typedef __attribute__((ext_vector_type(4)))  __bf16 v4bf;
typedef __attribute__((ext_vector_type(8)))  float  v8f;
typedef __attribute__((ext_vector_type(4)))  int    v4i;

constexpr int   kH     = 1024;              // hidden
constexpr int   kTok   = 2048;              // B*T
constexpr int   kV     = 32000;             // vocab
constexpr int   kSplit = 25;                // V splits per (model, token-group)
constexpr int   kVC    = kV / kSplit;       // 1280 cols per block
constexpr int   kVG    = kVC / 4;           // 320 cols per v-group (10 x 2 tiles)
constexpr int   kNP    = kSplit * 4;        // 100 partials per (model, token)
constexpr int   kFragT = 16 * kH;           // 16384 frag elems per 16-token tile
constexpr int   kIgn   = -100;
constexpr float kBeta  = 0.1f;

// ---- gfx1250 async global->LDS copy (guarded; fallback = vector LDS copy) --
// Builtin signature (from compiler diagnostic): (int4 AS1*, int4 AS3*, Ii, Ii)
#if __has_builtin(__builtin_amdgcn_global_load_async_to_lds_b128)
#define KTO_ASYNC_LDS 1
typedef __attribute__((address_space(1))) v4i g_v4i;   // global int4
typedef __attribute__((address_space(3))) v4i l_v4i;   // LDS int4
__device__ __forceinline__ g_v4i* kto_as1(const void* p) {
    return (g_v4i*)(unsigned long long)p;
}
__device__ __forceinline__ l_v4i* kto_as3(const void* p) {
    // generic LDS pointer: low 32 bits are the LDS offset (aperture layout)
    return (l_v4i*)(unsigned)(unsigned long long)p;
}
__device__ __forceinline__ void kto_wait_async0() {
#if __has_builtin(__builtin_amdgcn_s_wait_asynccnt)
    __builtin_amdgcn_s_wait_asynccnt(0);
#else
    asm volatile("s_wait_asynccnt 0x0" ::: "memory");
#endif
}
#else
#define KTO_ASYNC_LDS 0
#endif

// ------------------- kernel 0a: split W into bf16 hi/lo --------------------
__global__ __launch_bounds__(256)
void kto_wsplit(const float* __restrict__ W, const float* __restrict__ ref_W,
                __bf16* __restrict__ whi, __bf16* __restrict__ wlo) {
    const long per4 = (long)kV * kH / 4;                 // float4s per model
    const long g    = (long)blockIdx.x * 256 + threadIdx.x;
    const int  model = g >= per4;
    const long i4    = g - (model ? per4 : 0);
    const float4 f = ((const float4*)(model ? ref_W : W))[i4];

    v4bf h, l;
    h[0] = (__bf16)f.x;  l[0] = (__bf16)(f.x - (float)h[0]);
    h[1] = (__bf16)f.y;  l[1] = (__bf16)(f.y - (float)h[1]);
    h[2] = (__bf16)f.z;  l[2] = (__bf16)(f.z - (float)h[2]);
    h[3] = (__bf16)f.w;  l[3] = (__bf16)(f.w - (float)h[3]);

    const long b4 = (long)model * per4 + i4;             // in v4bf units
    ((v4bf*)whi)[b4] = h;
    ((v4bf*)wlo)[b4] = l;
}

// ------------------- kernel 0b: split x into fragment-order bf16 hi/lo -----
// Layout: [model][token-tile 0..127][frag 0..16383]; frag index
// (ks*32 + l)*16 + e holds x[tile*16 + (l&15)][ks*32 + 16*(l>>4) + e]
__global__ __launch_bounds__(256)
void kto_xsplit(const float* __restrict__ x, const float* __restrict__ ref_x,
                __bf16* __restrict__ xhi, __bf16* __restrict__ xlo) {
    const int idx   = blockIdx.x * 256 + threadIdx.x;    // < 2*128*16384 = 2^22
    const int f_    = idx & (kFragT - 1);
    const int tile  = (idx >> 14) & 127;
    const int model = idx >> 21;
    const int e  = f_ & 15;
    const int l  = (f_ >> 4) & 31;
    const int ks = f_ >> 9;
    const int m  = l & 15;
    const int k  = ks * 32 + 16 * (l >> 4) + e;
    const float* __restrict__ xp = model ? ref_x : x;
    const float f = xp[(long)(tile * 16 + m) * kH + k];
    const __bf16 h = (__bf16)f;
    xhi[idx] = h;
    xlo[idx] = (__bf16)(f - (float)h);
}

// ------------------- kernel 1: GEMM + per-lane online LSE ------------------
// Grid: 1600 blocks x 512 thr = 2 models x 32 token-groups x 25 V-splits.
// Block: 16 waves = 4 token-tiles x 4 v-groups; 64-token A tile in 256 KB LDS.
__global__ __launch_bounds__(512)
void kto_lse_partial(const __bf16* __restrict__ xhi, const __bf16* __restrict__ xlo,
                     const __bf16* __restrict__ whi, const __bf16* __restrict__ wlo,
                     float* __restrict__ pmax, float* __restrict__ psum) {
    extern __shared__ __bf16 sm[];                       // 256 KB dynamic LDS
    __bf16* s_hi = sm;                                   // [4 * kFragT]
    __bf16* s_lo = sm + 4 * kFragT;                      // [4 * kFragT]

    const int tid   = threadIdx.x;
    const int lane  = tid & 31;
    const int wv    = tid >> 5;                          // 0..15
    const int tm    = wv >> 2;                           // token tile 0..3
    const int tv    = wv & 3;                            // v-group 0..3
    const int blk   = blockIdx.x;
    const int grp   = blk & 31;                          // token group (64 tokens)
    const int model = (blk >> 5) & 1;
    const int split = blk >> 6;                          // 0..24

    // ---- stage pre-swizzled 4-tile x fragments (hi+lo, 128 KB each) into LDS
    const long xbase = ((long)model * 128 + grp * 4) * kFragT;   // 4 tiles contig
    const __bf16* __restrict__ ghi = xhi + xbase;
    const __bf16* __restrict__ glo = xlo + xbase;
#if KTO_ASYNC_LDS
    for (int ofs = tid * 8; ofs < 4 * kFragT; ofs += 512 * 8) {  // 16 B per lane
        __builtin_amdgcn_global_load_async_to_lds_b128(
            kto_as1(ghi + ofs), kto_as3(s_hi + ofs), 0, 0);
        __builtin_amdgcn_global_load_async_to_lds_b128(
            kto_as1(glo + ofs), kto_as3(s_lo + ofs), 0, 0);
    }
    kto_wait_async0();
    __syncthreads();
#else
    for (int ofs = tid * 8; ofs < 4 * kFragT; ofs += 512 * 8) {
        *(uint4*)&s_hi[ofs] = *(const uint4*)(ghi + ofs);
        *(uint4*)&s_lo[ofs] = *(const uint4*)(glo + ofs);
    }
    __syncthreads();
#endif

    const int half  = lane >> 4;
    const int nlan  = lane & 15;
    const int abase = tm * kFragT;

    // per-lane online logsumexp; C/D layout: lane holds rows m = r + 8*half.
    float Mrow[8], Srow[8];
#pragma unroll
    for (int r = 0; r < 8; ++r) { Mrow[r] = -INFINITY; Srow[r] = 0.0f; }

    const long mW = (long)model * kV * kH;
    const int  v0 = split * kVC + tv * kVG;

    for (int vt = 0; vt < kVG; vt += 32) {               // 10 passes x 2 N-tiles
        const int vb = v0 + vt;
        v8f acc0 = {}, acc1 = {};
        // B fragment: lane n=nlan reads 16 contiguous bf16 of one W row.
        const __bf16* pbh0 = whi + mW + (long)(vb + nlan) * kH + 16 * half;
        const __bf16* pbl0 = wlo + mW + (long)(vb + nlan) * kH + 16 * half;
        const __bf16* pbh1 = pbh0 + 16 * (long)kH;
        const __bf16* pbl1 = pbl0 + 16 * (long)kH;

        // pull the next pass's W rows toward L2/WGP$ (speculative; OOB dropped)
        __builtin_prefetch(pbh0 + 32 * (long)kH, 0, 0);
        __builtin_prefetch(pbl0 + 32 * (long)kH, 0, 0);

#pragma unroll 4
        for (int ks = 0; ks < 32; ++ks) {
            const int fb = abase + (ks * 32 + lane) * 16;
            const v16bf ah = *(const v16bf*)&s_hi[fb];
            const v16bf al = *(const v16bf*)&s_lo[fb];
            const v16bf bh0 = *(const v16bf*)(pbh0 + ks * 32);
            const v16bf bl0 = *(const v16bf*)(pbl0 + ks * 32);
            const v16bf bh1 = *(const v16bf*)(pbh1 + ks * 32);
            const v16bf bl1 = *(const v16bf*)(pbl1 + ks * 32);

            acc0 = __builtin_amdgcn_wmma_f32_16x16x32_bf16(false, al, false, bh0,
                                                           (short)0, acc0, false, false);
            acc0 = __builtin_amdgcn_wmma_f32_16x16x32_bf16(false, ah, false, bl0,
                                                           (short)0, acc0, false, false);
            acc0 = __builtin_amdgcn_wmma_f32_16x16x32_bf16(false, ah, false, bh0,
                                                           (short)0, acc0, false, false);
            acc1 = __builtin_amdgcn_wmma_f32_16x16x32_bf16(false, al, false, bh1,
                                                           (short)0, acc1, false, false);
            acc1 = __builtin_amdgcn_wmma_f32_16x16x32_bf16(false, ah, false, bl1,
                                                           (short)0, acc1, false, false);
            acc1 = __builtin_amdgcn_wmma_f32_16x16x32_bf16(false, ah, false, bh1,
                                                           (short)0, acc1, false, false);
        }

        // per-lane online update (no cross-lane traffic in steady state)
#pragma unroll
        for (int r = 0; r < 8; ++r) {
            float v  = acc0[r];
            float nM = fmaxf(Mrow[r], v);
            Srow[r]  = Srow[r] * __expf(Mrow[r] - nM) + __expf(v - nM);
            Mrow[r]  = nM;
            v  = acc1[r];
            nM = fmaxf(Mrow[r], v);
            Srow[r]  = Srow[r] * __expf(Mrow[r] - nM) + __expf(v - nM);
            Mrow[r]  = nM;
        }
    }

    // ---- single cross-lane (M,S) merge across the 16-lane N-group ----------
#pragma unroll
    for (int msk = 1; msk <= 8; msk <<= 1) {
#pragma unroll
        for (int r = 0; r < 8; ++r) {
            const float Mo = __shfl_xor(Mrow[r], msk, 32);
            const float So = __shfl_xor(Srow[r], msk, 32);
            const float nM = fmaxf(Mrow[r], Mo);
            Srow[r] = Srow[r] * __expf(Mrow[r] - nM) + So * __expf(Mo - nM);
            Mrow[r] = nM;
        }
    }

    if (nlan == 0) {   // lanes 0 / 16 write rows 0-7 / 8-15
        const int  P    = split * 4 + tv;
        const long base = ((long)model * kNP + P) * kTok
                        + (long)grp * 64 + tm * 16 + 8 * half;
#pragma unroll
        for (int r = 0; r < 8; ++r) {
            pmax[base + r] = Mrow[r];
            psum[base + r] = Srow[r];
        }
    }
}

// ------------------- kernel 2: label-logit dot -----------------------------
__global__ __launch_bounds__(256)
void kto_label_dot(const float* __restrict__ x, const float* __restrict__ ref_x,
                   const float* __restrict__ W, const float* __restrict__ ref_W,
                   const int* __restrict__ y, float* __restrict__ lab) {
    const int lane  = threadIdx.x & 31;
    const int wid   = blockIdx.x * 8 + (threadIdx.x >> 5);
    const int tok   = wid & (kTok - 1);
    const int model = wid >> 11;
    const float* __restrict__ xp = model ? ref_x : x;
    const float* __restrict__ wp = model ? ref_W : W;

    const int label = y[tok];
    float s = 0.0f;
    if (label != kIgn) {
        const float* xr = xp + (long)tok * kH;
        const float* wr = wp + (long)label * kH;
#pragma unroll 4
        for (int h = lane; h < kH; h += 32)
            s = fmaf(xr[h], wr[h], s);
    }
#pragma unroll
    for (int m = 16; m >= 1; m >>= 1) s += __shfl_xor(s, m, 32);
    if (lane == 0) lab[model * kTok + tok] = s;
}

// ------------------- kernel 3a: combine partials -> per-token logp ---------
__global__ __launch_bounds__(256)
void kto_combine(const float* __restrict__ pmax, const float* __restrict__ psum,
                 const float* __restrict__ lab, const int* __restrict__ y,
                 float* __restrict__ tokl) {
    const int gid   = blockIdx.x * 256 + threadIdx.x;
    const int tok   = gid & (kTok - 1);
    const int model = gid >> 11;
    if (y[tok] == kIgn) { tokl[gid] = 0.0f; return; }

    float M = -INFINITY;
    for (int p = 0; p < kNP; ++p)
        M = fmaxf(M, pmax[((long)model * kNP + p) * kTok + tok]);
    float S = 0.0f;
    for (int p = 0; p < kNP; ++p) {
        const long i = ((long)model * kNP + p) * kTok + tok;
        S += psum[i] * __expf(pmax[i] - M);
    }
    tokl[gid] = lab[gid] - (M + __logf(S));
}

// ------------------- kernel 3b: masked means + KTO loss --------------------
__global__ __launch_bounds__(256)
void kto_finalize(const float* __restrict__ tokl, const int* __restrict__ y,
                  const unsigned char* __restrict__ pref, const float* __restrict__ kl,
                  float* __restrict__ out) {
    __shared__ float s_sum[4][256];   // (model, b)
    __shared__ float s_cnt[2][256];
    const int tid = threadIdx.x;

    float acc[4] = {0.f, 0.f, 0.f, 0.f};
    float cnt[2] = {0.f, 0.f};
    for (int tok = tid; tok < kTok; tok += 256) {
        const int b = tok >> 10;
        if (y[tok] != kIgn) {
            cnt[b] += 1.0f;
            acc[0 * 2 + b] += tokl[0 * kTok + tok];
            acc[1 * 2 + b] += tokl[1 * kTok + tok];
        }
    }
#pragma unroll
    for (int i = 0; i < 4; ++i) s_sum[i][tid] = acc[i];
    s_cnt[0][tid] = cnt[0];
    s_cnt[1][tid] = cnt[1];
    __syncthreads();
    for (int stp = 128; stp >= 1; stp >>= 1) {       // fixed order: deterministic
        if (tid < stp) {
#pragma unroll
            for (int i = 0; i < 4; ++i) s_sum[i][tid] += s_sum[i][tid + stp];
            s_cnt[0][tid] += s_cnt[0][tid + stp];
            s_cnt[1][tid] += s_cnt[1][tid + stp];
        }
        __syncthreads();
    }
    if (tid == 0) {
        const float klv = kl[0];
        float total = 0.0f;
        for (int b = 0; b < 2; ++b) {
            const float pol = s_sum[0 * 2 + b][0] / s_cnt[b][0];
            const float rfl = s_sum[1 * 2 + b][0] / s_cnt[b][0];
            const float lr  = pol - rfl;
            const float chosen   = 1.0f - 1.0f / (1.0f + __expf(-(kBeta * (lr - klv))));
            const float rejected = 1.0f - 1.0f / (1.0f + __expf(-(kBeta * (klv - lr))));
            total += pref[b] ? chosen : rejected;
        }
        out[0] = 0.5f * total;
    }
}

// ---------------------------------------------------------------------------
extern "C" void kernel_launch(void* const* d_in, const int* in_sizes, int n_in,
                              void* d_out, int out_size, void* d_ws, size_t ws_size,
                              hipStream_t stream) {
    const float*         x     = (const float*)d_in[0];
    const float*         ref_x = (const float*)d_in[1];
    const float*         W     = (const float*)d_in[2];
    const float*         ref_W = (const float*)d_in[3];
    const int*           y     = (const int*)d_in[4];
    const unsigned char* pref  = (const unsigned char*)d_in[5];
    const float*         kl    = (const float*)d_in[6];
    float*               out   = (float*)d_out;

    // workspace layout (floats first, then bf16 arrays; all 16B aligned)
    float* pmax = (float*)d_ws;                            // [2*kNP*kTok]
    float* psum = pmax + 2 * kNP * kTok;                   // [2*kNP*kTok]
    float* lab  = psum + 2 * kNP * kTok;                   // [2*kTok]
    float* tokl = lab + 2 * kTok;                          // [2*kTok]
    __bf16* whi = (__bf16*)(tokl + 2 * kTok);              // [2*kV*kH]
    __bf16* wlo = whi + (long)2 * kV * kH;                 // [2*kV*kH]
    __bf16* xhi = wlo + (long)2 * kV * kH;                 // [2*kTok*kH]
    __bf16* xlo = xhi + (long)2 * kTok * kH;               // [2*kTok*kH]

    // 0a) split W/ref_W into bf16 hi/lo: 2*V*H/4 float4s / 256 = 64000 blocks
    kto_wsplit<<<64000, 256, 0, stream>>>(W, ref_W, whi, wlo);
    // 0b) split x/ref_x into fragment-order bf16 hi/lo: 2^22 / 256 = 16384 blocks
    kto_xsplit<<<16384, 256, 0, stream>>>(x, ref_x, xhi, xlo);
    // 1) main GEMM + online LSE: 2 x 32 x 25 = 1600 blocks, 256 KB dynamic LDS
    kto_lse_partial<<<1600, 512, 8 * kFragT * (int)sizeof(__bf16), stream>>>(
        xhi, xlo, whi, wlo, pmax, psum);
    // 2) label logits: 4096 waves -> 512 blocks
    kto_label_dot<<<512, 256, 0, stream>>>(x, ref_x, W, ref_W, y, lab);
    // 3) combine + finalize
    kto_combine<<<16, 256, 0, stream>>>(pmax, psum, lab, y, tokl);
    kto_finalize<<<1, 256, 0, stream>>>(tokl, y, pref, kl, out);
}
